// CausalSelfAttention_20736102105352
// MI455X (gfx1250) — matrix operand order repaired and checked by reading, not compiled, not run on hardware
//
#include <hip/hip_runtime.h>
#include <hip/hip_bf16.h>

// ---------------------------------------------------------------------------
// CausalSelfAttention forward for MI455X (gfx1250, wave32, WMMA f32_16x16x32_f16)
// B=4, T=2048, C_IN=1152, N_EMBD=1024, H=16, hd=64
// ---------------------------------------------------------------------------

#define B_    4
#define T_    2048
#define CIN_  1152
#define NE_   1024
#define H_    16
#define HD_   64
#define N3_   3072
#define ROWS_ (B_ * T_)   // 8192

typedef __attribute__((ext_vector_type(16))) _Float16 v16h;
typedef __attribute__((ext_vector_type(8)))  float    v8f;

union F8 { v8f v; float f[8]; };

__device__ __forceinline__ v8f wmma_f16(v16h a, v16h b, v8f c) {
  // 8 args: (neg_a, A, neg_b, B, c_mod, C, reuse_a, reuse_b)
  return __builtin_amdgcn_wmma_f32_16x16x32_f16(false, a, false, b, (short)0, c,
                                                false, false);
}

// A-operand fragment (16x32 f16):
//   lane<16: row=lane,    K=0..15 contiguous  (p = rowBase+kc+0)
//   lane>=16: row=lane-16, K=16..31 contiguous (p = rowBase+kc+16)
__device__ __forceinline__ v16h load_fragA(const _Float16* p) {
  union { v16h v; uint4 u[2]; } t;
  t.u[0] = *reinterpret_cast<const uint4*>(p);
  t.u[1] = *reinterpret_cast<const uint4*>(p + 8);
  return t.v;
}

// B-operand fragment (32x16 f16): lane n holds column n%16;
//   lane<16: K=0..15 contiguous; lane>=16: K=16..31 contiguous.
//   caller passes p = colBase + kc + (hi ? 16 : 0)
__device__ __forceinline__ v16h load_fragB(const _Float16* p) {
  union { v16h v; uint4 u[2]; } t;
  t.u[0] = *reinterpret_cast<const uint4*>(p);
  t.u[1] = *reinterpret_cast<const uint4*>(p + 8);
  return t.v;
}

// ---------------------------------------------------------------------------
// 1) LayerNorm + convert to f16.  One block per row of 1152.
// ---------------------------------------------------------------------------
__global__ void __launch_bounds__(256)
ln_to_f16(const float* __restrict__ x, const float* __restrict__ w,
          const float* __restrict__ b, _Float16* __restrict__ h) {
  const int row = blockIdx.x;
  const float* xr = x + (size_t)row * CIN_;
  float s = 0.f, s2 = 0.f;
  for (int c = threadIdx.x; c < CIN_; c += 256) {
    float v = xr[c];
    s += v; s2 += v * v;
  }
  __shared__ float r1[256], r2[256];
  r1[threadIdx.x] = s; r2[threadIdx.x] = s2;
  __syncthreads();
  for (int o = 128; o > 0; o >>= 1) {
    if (threadIdx.x < o) {
      r1[threadIdx.x] += r1[threadIdx.x + o];
      r2[threadIdx.x] += r2[threadIdx.x + o];
    }
    __syncthreads();
  }
  const float mu  = r1[0] * (1.0f / CIN_);
  const float var = r2[0] * (1.0f / CIN_) - mu * mu;
  const float rs  = rsqrtf(var + 1e-5f);
  for (int c = threadIdx.x; c < CIN_; c += 256) {
    float v = (xr[c] - mu) * rs * w[c] + b[c];
    h[(size_t)row * CIN_ + c] = (_Float16)v;
  }
}

// ---------------------------------------------------------------------------
// 2) Transpose + convert weights: W[K,N] f32 -> Wt[N,K] f16
// ---------------------------------------------------------------------------
__global__ void __launch_bounds__(256)
conv_transpose(const float* __restrict__ W, _Float16* __restrict__ Wt,
               int K, int N) {
  int idx = blockIdx.x * 256 + threadIdx.x;
  if (idx >= K * N) return;
  int k = idx / N, n = idx % N;
  Wt[(size_t)n * K + k] = (_Float16)W[idx];
}

// ---------------------------------------------------------------------------
// 3) WMMA GEMM: C[M,N] = A[M,K](f16) * Bt[N,K]^T(f16) + bias, OT = f16 or f32
//    Block tile 128x128, 8 waves (2M x 4N), wave tile 64x32 -> 8 WMMA / K-step
//    K-loop is double-buffered in registers: next tile's 12 b128 loads are in
//    flight while the current 8 WMMAs execute.
// ---------------------------------------------------------------------------
template <typename OT>
__global__ void __launch_bounds__(256)
gemm_wmma(const _Float16* __restrict__ A, const _Float16* __restrict__ Bt,
          const float* __restrict__ bias, OT* __restrict__ C,
          int M, int N, int K) {
  const int lane = threadIdx.x & 31;
  const int wave = threadIdx.x >> 5;
  const int l16  = lane & 15;
  const int hi   = lane >> 4;
  const int m0   = blockIdx.x * 128 + (wave & 1) * 64;
  const int n0   = blockIdx.y * 128 + (wave >> 1) * 32;
  const int koffA = hi * 16;
  const int koffB = hi * 16;

  F8 acc[4][2];
#pragma unroll
  for (int ms = 0; ms < 4; ++ms)
#pragma unroll
    for (int ns = 0; ns < 2; ++ns)
#pragma unroll
      for (int r = 0; r < 8; ++r) acc[ms][ns].f[r] = 0.f;

  v16h af[2][4], bf[2][2];
  // prologue: load K-step 0 into buffer 0
#pragma unroll
  for (int ms = 0; ms < 4; ++ms)
    af[0][ms] = load_fragA(A + (size_t)(m0 + ms * 16 + l16) * K + koffA);
#pragma unroll
  for (int ns = 0; ns < 2; ++ns)
    bf[0][ns] = load_fragB(Bt + (size_t)(n0 + ns * 16 + l16) * K + koffB);

  int buf = 0;
  for (int kc = 0; kc < K; kc += 32) {
    const int kn = kc + 32;
    if (kn < K) {  // issue next tile's loads before consuming current tile
      if (kn + 32 < K) {  // and hint the tile after that: global_prefetch_b8
        __builtin_prefetch(A + (size_t)(m0 + l16) * K + kn + 32, 0, 0);
        __builtin_prefetch(Bt + (size_t)(n0 + l16) * K + kn + 32, 0, 0);
      }
#pragma unroll
      for (int ms = 0; ms < 4; ++ms)
        af[buf ^ 1][ms] =
            load_fragA(A + (size_t)(m0 + ms * 16 + l16) * K + kn + koffA);
#pragma unroll
      for (int ns = 0; ns < 2; ++ns)
        bf[buf ^ 1][ns] =
            load_fragB(Bt + (size_t)(n0 + ns * 16 + l16) * K + kn + koffB);
    }
#pragma unroll
    for (int ms = 0; ms < 4; ++ms)
#pragma unroll
      for (int ns = 0; ns < 2; ++ns)
        acc[ms][ns].v = wmma_f16(af[buf][ms], bf[buf][ns], acc[ms][ns].v);
    buf ^= 1;
  }

  // C/D layout: VGPR r, lane<16 -> M=r; lane>=16 -> M=8+r; N = l16
#pragma unroll
  for (int ns = 0; ns < 2; ++ns) {
    const int col = n0 + ns * 16 + l16;
    const float bv = bias[col];
#pragma unroll
    for (int ms = 0; ms < 4; ++ms)
#pragma unroll
      for (int r = 0; r < 8; ++r) {
        const int m = m0 + ms * 16 + r + 8 * hi;
        C[(size_t)m * N + col] = (OT)(acc[ms][ns].f[r] + bv);
      }
  }
}

// ---------------------------------------------------------------------------
// 4) Pack V transposed: qkv[B,T,3072] f16 -> vT[B,H,hd,T] f16
// ---------------------------------------------------------------------------
__global__ void __launch_bounds__(256)
pack_v(const _Float16* __restrict__ qkv, _Float16* __restrict__ vT) {
  size_t idx = (size_t)blockIdx.x * 256 + threadIdx.x;  // over B*T*NE
  if (idx >= (size_t)B_ * T_ * NE_) return;
  const int c = (int)(idx & (NE_ - 1));
  const size_t bt = idx >> 10;         // b*T + t
  const int t = (int)(bt & (T_ - 1));
  const int b = (int)(bt >> 11);
  const int h = c >> 6, d = c & (HD_ - 1);
  vT[(((size_t)(b * H_ + h) * HD_ + d) << 11) + t] =
      qkv[bt * N3_ + 2 * NE_ + c];
}

// ---------------------------------------------------------------------------
// 5) Flash attention (causal). Block = 128 q rows of one (b,h); 8 waves x 16.
//    Key tiles of 64 staged block-wide into LDS (8x reuse across waves).
//    Online softmax in f32, WMMA f16 for QK^T and PV.
// ---------------------------------------------------------------------------
#define LP_ 72   // padded LDS row stride (halfs): 144B, spreads banks
__global__ void __launch_bounds__(256)
attn_flash(const _Float16* __restrict__ qkv, const _Float16* __restrict__ vT,
           _Float16* __restrict__ y) {
  const int lane = threadIdx.x & 31;
  const int wave = threadIdx.x >> 5;
  const int l16  = lane & 15;
  const int hi   = lane >> 4;
  const int qt = blockIdx.x;   // T/128
  const int h  = blockIdx.y;   // heads
  const int b  = blockIdx.z;   // batch
  const int q0 = qt * 128 + wave * 16;
  const int koffA = hi * 16;
  const int koffB = hi * 16;

  __shared__ _Float16 Klds[64][LP_];      // [key][hd]  tile
  __shared__ _Float16 Vlds[64][LP_];      // [hd][key]  tile (from vT)
  __shared__ _Float16 plds[8][16][LP_];   // wave-private P relayout C->A
  _Float16(*pw)[LP_] = plds[wave];

  const _Float16* qbase = qkv + (size_t)(b * T_) * N3_ + h * HD_;
  const _Float16* kbase = qbase + NE_;
  const _Float16* vbase = vT + (size_t)(b * H_ + h) * HD_ * T_;

  // Q fragments for this wave's 16 rows (loaded once)
  v16h qf[2];
#pragma unroll
  for (int ks = 0; ks < 2; ++ks)
    qf[ks] = load_fragA(qbase + (size_t)(q0 + l16) * N3_ + ks * 32 + koffA);

  F8 o[4];
  float mrow[8], lrow[8];
#pragma unroll
  for (int n = 0; n < 4; ++n)
#pragma unroll
    for (int r = 0; r < 8; ++r) o[n].f[r] = 0.f;
#pragma unroll
  for (int r = 0; r < 8; ++r) { mrow[r] = -3.0e38f; lrow[r] = 0.f; }

  const float scale = 0.125f;  // 1/sqrt(64)
  const int ktmax = (qt * 128 + 127) >> 6;  // block-uniform causal bound

  for (int kt = 0; kt <= ktmax; ++kt) {
    // ---- cooperative staging of K,V tiles (coalesced b128 -> ds b128)
    __syncthreads();
    {
      // 64 rows x 8 chunks of 16B each, 512 chunks over 256 threads
#pragma unroll
      for (int i = 0; i < 2; ++i) {
        const int chunk = threadIdx.x + i * 256;
        const int row = chunk >> 3;
        const int seg = (chunk & 7) * 8;  // half offset
        *reinterpret_cast<uint4*>(&Klds[row][seg]) =
            *reinterpret_cast<const uint4*>(
                kbase + (size_t)(kt * 64 + row) * N3_ + seg);
        *reinterpret_cast<uint4*>(&Vlds[row][seg]) =
            *reinterpret_cast<const uint4*>(
                vbase + (size_t)row * T_ + kt * 64 + seg);
      }
    }
    __syncthreads();

    if (kt * 64 > q0 + 15) continue;  // wave-uniform: fully masked tile

    // ---- S = Q @ K^T : load all 8 fragments first, then 8 WMMA
    v16h kf[4][2];
#pragma unroll
    for (int n = 0; n < 4; ++n)
#pragma unroll
      for (int ks = 0; ks < 2; ++ks)
        kf[n][ks] = load_fragB(&Klds[n * 16 + l16][ks * 32 + koffB]);

    F8 s[4];
#pragma unroll
    for (int n = 0; n < 4; ++n) {
#pragma unroll
      for (int r = 0; r < 8; ++r) s[n].f[r] = 0.f;
      s[n].v = wmma_f16(qf[0], kf[n][0], s[n].v);
      s[n].v = wmma_f16(qf[1], kf[n][1], s[n].v);
    }
    // ---- scale + causal mask
#pragma unroll
    for (int n = 0; n < 4; ++n) {
      const int key = kt * 64 + n * 16 + l16;
#pragma unroll
      for (int r = 0; r < 8; ++r) {
        const int qr = q0 + r + 8 * hi;
        const float v = s[n].f[r] * scale;
        s[n].f[r] = (key <= qr) ? v : -3.0e38f;
      }
    }
    // ---- online softmax per row (rows live in lane-half x VGPR r)
#pragma unroll
    for (int r = 0; r < 8; ++r) {
      float tm = fmaxf(fmaxf(s[0].f[r], s[1].f[r]),
                       fmaxf(s[2].f[r], s[3].f[r]));
      tm = fmaxf(tm, __shfl_xor(tm, 1, 32));
      tm = fmaxf(tm, __shfl_xor(tm, 2, 32));
      tm = fmaxf(tm, __shfl_xor(tm, 4, 32));
      tm = fmaxf(tm, __shfl_xor(tm, 8, 32));
      const float mn = fmaxf(mrow[r], tm);
      const float alpha = __expf(mrow[r] - mn);
      float ts = 0.f;
#pragma unroll
      for (int n = 0; n < 4; ++n) {
        const float p = __expf(s[n].f[r] - mn);
        s[n].f[r] = p;
        ts += p;
      }
      ts += __shfl_xor(ts, 1, 32);
      ts += __shfl_xor(ts, 2, 32);
      ts += __shfl_xor(ts, 4, 32);
      ts += __shfl_xor(ts, 8, 32);
      lrow[r] = lrow[r] * alpha + ts;
      mrow[r] = mn;
#pragma unroll
      for (int n = 0; n < 4; ++n) o[n].f[r] *= alpha;
    }
    // ---- P: C-layout regs -> LDS -> A-layout fragments (wave-private)
#pragma unroll
    for (int n = 0; n < 4; ++n)
#pragma unroll
      for (int r = 0; r < 8; ++r)
        pw[r + 8 * hi][n * 16 + l16] = (_Float16)s[n].f[r];

    v16h pf[2];
#pragma unroll
    for (int ks = 0; ks < 2; ++ks) {
      union { v16h v; uint4 u[2]; } t;
      const _Float16* pp = &pw[l16][ks * 32 + koffA];
      t.u[0] = *reinterpret_cast<const uint4*>(pp);
      t.u[1] = *reinterpret_cast<const uint4*>(pp + 8);
      pf[ks] = t.v;
    }
    // ---- O += P @ V : load all 8 V fragments first, then 8 WMMA
    v16h vf[4][2];
#pragma unroll
    for (int n = 0; n < 4; ++n)
#pragma unroll
      for (int ks = 0; ks < 2; ++ks)
        vf[n][ks] = load_fragB(&Vlds[n * 16 + l16][ks * 32 + koffB]);
#pragma unroll
    for (int n = 0; n < 4; ++n) {
      o[n].v = wmma_f16(pf[0], vf[n][0], o[n].v);
      o[n].v = wmma_f16(pf[1], vf[n][1], o[n].v);
    }
  }

  // ---- normalize and store y (f16, [B,T,NE] with head offset)
#pragma unroll
  for (int n = 0; n < 4; ++n) {
    const int col = h * HD_ + n * 16 + l16;
#pragma unroll
    for (int r = 0; r < 8; ++r) {
      const int qr = q0 + r + 8 * hi;
      const float val = o[n].f[r] / lrow[r];
      y[(size_t)(b * T_ + qr) * NE_ + col] = (_Float16)val;
    }
  }
}

// ---------------------------------------------------------------------------
// Launch
// ---------------------------------------------------------------------------
extern "C" void kernel_launch(void* const* d_in, const int* in_sizes, int n_in,
                              void* d_out, int out_size, void* d_ws,
                              size_t ws_size, hipStream_t stream) {
  const float* x      = (const float*)d_in[0];
  const float* ln_w   = (const float*)d_in[1];
  const float* ln_b   = (const float*)d_in[2];
  const float* W_attn = (const float*)d_in[3];
  const float* b_attn = (const float*)d_in[4];
  const float* W_proj = (const float*)d_in[5];
  const float* b_proj = (const float*)d_in[6];
  float* out = (float*)d_out;

  _Float16* ws = (_Float16*)d_ws;
  _Float16* h_f16  = ws;                                  // ROWS_*CIN_
  _Float16* Wa_t   = h_f16 + (size_t)ROWS_ * CIN_;        // N3_*CIN_
  _Float16* qkv    = Wa_t + (size_t)N3_ * CIN_;           // ROWS_*N3_
  _Float16* vT     = qkv + (size_t)ROWS_ * N3_;           // B_*H_*HD_*T_
  _Float16* y_f16  = vT + (size_t)B_ * H_ * HD_ * T_;     // ROWS_*NE_
  _Float16* Wp_t   = y_f16 + (size_t)ROWS_ * NE_;         // NE_*NE_

  // 1) LayerNorm -> f16
  ln_to_f16<<<ROWS_, 256, 0, stream>>>(x, ln_w, ln_b, h_f16);

  // 2) weight transpose+convert
  conv_transpose<<<(CIN_ * N3_ + 255) / 256, 256, 0, stream>>>(W_attn, Wa_t,
                                                               CIN_, N3_);
  conv_transpose<<<(NE_ * NE_ + 255) / 256, 256, 0, stream>>>(W_proj, Wp_t,
                                                              NE_, NE_);

  // 3) QKV GEMM (f16 out): [8192,3072] = h @ W_attn + b_attn
  gemm_wmma<_Float16><<<dim3(ROWS_ / 128, N3_ / 128), 256, 0, stream>>>(
      h_f16, Wa_t, b_attn, qkv, ROWS_, N3_, CIN_);

  // 4) pack V transposed
  pack_v<<<(int)(((size_t)B_ * T_ * NE_ + 255) / 256), 256, 0, stream>>>(qkv,
                                                                         vT);

  // 5) flash attention
  attn_flash<<<dim3(T_ / 128, H_, B_), 256, 0, stream>>>(qkv, vT, y_f16);

  // 6) projection GEMM (f32 out): out = y @ W_proj + b_proj
  gemm_wmma<float><<<dim3(ROWS_ / 128, NE_ / 128), 256, 0, stream>>>(
      y_f16, Wp_t, b_proj, out, ROWS_, NE_, NE_);
}
